// RoutedMoE_87686052315536
// MI455X (gfx1250) — matrix operand order repaired and checked by reading, not compiled, not run on hardware
//
#include <hip/hip_runtime.h>
#include <math.h>
#include <stdint.h>

// ---------------- problem constants (from reference) ----------------
#define B_SZ   2
#define S_SZ   512
#define H_DIM  1024
#define E_NUM  64
#define I_DIM  512
#define K_TOP  8
#define G_NUM  8
#define TG_NUM 4
#define T_TOK  (B_SZ * S_SZ)      // 1024
#define CAPACITY 512
#define SCALE_F 2.5f
#define ROWS_MAX (T_TOK * K_TOP + T_TOK)  // 9216 compacted rows incl. shared expert

// ---------------- CDNA5 WMMA types ----------------
typedef unsigned short ushort_t;
typedef __bf16 bf16_t;
typedef bf16_t v16bf __attribute__((ext_vector_type(16)));
typedef bf16_t v2bf  __attribute__((ext_vector_type(2)));
typedef float  v8f   __attribute__((ext_vector_type(8)));
typedef int    v2i_t __attribute__((ext_vector_type(2)));

#define AS1 __attribute__((address_space(1)))
#define AS3 __attribute__((address_space(3)))

#if __has_builtin(__builtin_amdgcn_cvt_pk_bf16_f32)
#define HAVE_CVT_PK_BF16 1
#else
#define HAVE_CVT_PK_BF16 0
#endif

#if __has_builtin(__builtin_amdgcn_global_load_async_to_lds_b64) && \
    __has_builtin(__builtin_amdgcn_s_wait_asynccnt)
#define HAVE_ASYNC_LDS 1
#else
#define HAVE_ASYNC_LDS 0
#endif

union FragU { unsigned u32[8]; v16bf b; };

__device__ __forceinline__ unsigned pack2bf(float lo, float hi) {
#if HAVE_CVT_PK_BF16
    v2bf p = __builtin_amdgcn_cvt_pk_bf16_f32(lo, hi);
    return __builtin_bit_cast(unsigned, p);
#else
    unsigned ul = __builtin_bit_cast(unsigned, lo);
    unsigned uh = __builtin_bit_cast(unsigned, hi);
    ul += 0x7FFFu + ((ul >> 16) & 1u);
    uh += 0x7FFFu + ((uh >> 16) & 1u);
    return (ul >> 16) | (uh & 0xFFFF0000u);
#endif
}
__device__ __forceinline__ ushort_t f2bf(float f) {
    return (ushort_t)pack2bf(f, f);
}

__device__ __forceinline__ v8f wmma_bf16(v16bf a, v16bf b, v8f c) {
    // D = A(16x32) * B(32x16) + C(16x16 f32)
    return __builtin_amdgcn_wmma_f32_16x16x32_bf16(
        false, a, false, b, (short)0, c, false, false);
}

// A-fragment (16x32 bf16): lane<16 holds row M=lane, K = {0..7,16..23};
// lane>=16 holds row M=lane-16, K = {8..15,24..31}.  (ISA 7.12.2)
__device__ __forceinline__ v16bf load_a_frag(const ushort_t* As, int m, int half) {
    FragU f;
    const ushort_t* row = As + m * 36;
#pragma unroll
    for (int i = 0; i < 8; ++i) {
        const int k2 = 2 * i + (i >= 4 ? 8 : 0) + 8 * half;
        f.u32[i] = *(const unsigned*)(row + k2);
    }
    return f.b;
}

// B-fragment (32x16 bf16): lane<16 -> col N=lane, K = {0..7,16..23}; lane>=16 -> K = {8..15,24..31}.
// LDS holds B transposed: Bs[n][k].
__device__ __forceinline__ v16bf load_b_frag(const ushort_t* Bs, int n, int half) {
    FragU f;
    const ushort_t* row = Bs + n * 36;
#pragma unroll
    for (int i = 0; i < 8; ++i) f.u32[i] = *(const unsigned*)(row + 2 * i + (i >= 4 ? 8 : 0) + 8 * half);
    return f.b;
}

// ============================================================================
// Kernel 1: router — logits, sigmoid, grouped top-k, weights
// ============================================================================
__global__ void router_kernel(const float* __restrict__ x,
                              const float* __restrict__ rw,
                              int* __restrict__ topk_idx,
                              float* __restrict__ topk_w) {
    const int t = blockIdx.x;
    const int e = threadIdx.x;             // 0..63
    __shared__ float sc[E_NUM];

    const float* xt = x + (size_t)t * H_DIM;
    float acc = 0.f;
#pragma unroll 4
    for (int k = 0; k < H_DIM; ++k)
        acc = fmaf(xt[k], rw[(size_t)k * E_NUM + e], acc);
    sc[e] = 1.0f / (1.0f + expf(-acc));
    __syncthreads();

    if (e == 0) {
        float gs[G_NUM];
#pragma unroll
        for (int g = 0; g < G_NUM; ++g) {
            float m = sc[g * 8];
#pragma unroll
            for (int j = 1; j < 8; ++j) m = fmaxf(m, sc[g * 8 + j]);
            gs[g] = m;
        }
        bool gsel[G_NUM];
        for (int g = 0; g < G_NUM; ++g) gsel[g] = false;
        for (int it = 0; it < TG_NUM; ++it) {
            int bi = -1; float bv = -1.0f;
            for (int g = 0; g < G_NUM; ++g)
                if (!gsel[g] && gs[g] > bv) { bv = gs[g]; bi = g; }
            gsel[bi] = true;
        }
        float ms[E_NUM];
        for (int i = 0; i < E_NUM; ++i) ms[i] = gsel[i >> 3] ? sc[i] : 0.0f;
        int   sel[K_TOP];
        float w[K_TOP];
        bool  used[E_NUM];
        for (int i = 0; i < E_NUM; ++i) used[i] = false;
        float sum = 0.f;
        for (int it = 0; it < K_TOP; ++it) {
            int bi = -1; float bv = -1.0f;
            for (int i = 0; i < E_NUM; ++i)
                if (!used[i] && ms[i] > bv) { bv = ms[i]; bi = i; }
            used[bi] = true; sel[it] = bi; w[it] = bv; sum += bv;
        }
        const float s = SCALE_F / fmaxf(sum, 1e-9f);
        for (int it = 0; it < K_TOP; ++it) {
            topk_idx[t * K_TOP + it] = sel[it];
            topk_w[t * K_TOP + it]   = w[it] * s;
        }
    }
}

// ============================================================================
// Kernel 2: dispatch — reference slot order + compacted row ranges
// ============================================================================
__global__ void dispatch_kernel(const int* __restrict__ topk_idx,
                                int* __restrict__ assign_slot,
                                int* __restrict__ tok_list,
                                int* __restrict__ row_base_g) {
    const int tid = threadIdx.x;           // 128 threads
    __shared__ int cnt_s[E_NUM + 1];
    __shared__ int rb_s[E_NUM + 1];

    if (tid < E_NUM) {
        int c = 0;
        for (int a = 0; a < T_TOK * K_TOP; ++a)
            if (topk_idx[a] == tid) ++c;
        cnt_s[tid] = c < CAPACITY ? c : CAPACITY;
    } else if (tid == E_NUM) {
        cnt_s[E_NUM] = T_TOK;              // shared expert pseudo-group
    }
    __syncthreads();
    if (tid == 0) {
        int s = 0;
        for (int g = 0; g <= E_NUM; ++g) { rb_s[g] = s; s += cnt_s[g]; }
    }
    __syncthreads();
    if (tid <= E_NUM) {
        row_base_g[tid] = rb_s[tid];
        row_base_g[E_NUM + 1 + tid] = cnt_s[tid];   // counts stored after bases
    }
    if (tid < E_NUM) {
        int c = 0; const int rb = rb_s[tid];
        for (int a = 0; a < T_TOK * K_TOP; ++a) {
            if (topk_idx[a] == tid) {
                if (c < CAPACITY) {
                    assign_slot[a] = c;
                    tok_list[rb + c] = a / K_TOP;
                } else {
                    assign_slot[a] = -1;    // overflow -> dropped (valid=0)
                }
                ++c;
            }
        }
    }
    const int sb = rb_s[E_NUM];
    for (int r = tid; r < T_TOK; r += blockDim.x) tok_list[sb + r] = r;
}

// ============================================================================
// Kernel 3: fused gate/up grouped GEMM + SwiGLU epilogue (software pipelined)
// ============================================================================
__global__ __launch_bounds__(256) void gateup_kernel(
    const float* __restrict__ x,
    const float* __restrict__ gate_w, const float* __restrict__ up_w,
    const float* __restrict__ sgate_w, const float* __restrict__ sup_w,
    const int* __restrict__ tok_list, const int* __restrict__ row_base_g,
    ushort_t* __restrict__ h_bf) {
    const int g    = blockIdx.z;
    const int n0   = blockIdx.x * 64;
    const int m0   = blockIdx.y * 128;
    const int rows = row_base_g[E_NUM + 1 + g];
    if (m0 >= rows) return;
    const int rbase = row_base_g[g];
    const float* Wg = (g < E_NUM) ? gate_w + (size_t)g * H_DIM * I_DIM : sgate_w;
    const float* Wu = (g < E_NUM) ? up_w   + (size_t)g * H_DIM * I_DIM : sup_w;

    __shared__ __align__(16) ushort_t As[128 * 36];
    __shared__ __align__(16) ushort_t Bg[64 * 36];
    __shared__ __align__(16) ushort_t Bu[64 * 36];

    const int tid  = threadIdx.x;
    const int lane = tid & 31;
    const int wv   = tid >> 5;
    const int wm   = (wv & 3) * 32;
    const int wn   = (wv >> 2) * 32;
    const int half = lane >> 4;
    const int l16  = lane & 15;

    // ---- K-invariant load geometry (hoisted; branchless via row clamp) ----
    int mA[4], kkA[4];
    const float* pA[4];
#pragma unroll
    for (int li = 0; li < 4; ++li) {
        const int idx = tid + li * 256;           // 0..1023
        mA[li]  = idx >> 3;
        kkA[li] = (idx & 7) * 4;
        int row = m0 + mA[li]; row = row < rows ? row : rows - 1;
        pA[li] = x + (size_t)tok_list[rbase + row] * H_DIM + kkA[li];
    }
    int kkB[2], nB[2];
    const float* pBg[2];
    const float* pBu[2];
#pragma unroll
    for (int li = 0; li < 2; ++li) {
        const int idx = tid + li * 256;           // 0..511
        kkB[li] = idx >> 4;
        nB[li]  = (idx & 15) * 4;
        pBg[li] = Wg + (size_t)kkB[li] * I_DIM + n0 + nB[li];
        pBu[li] = Wu + (size_t)kkB[li] * I_DIM + n0 + nB[li];
    }

    v8f accg[2][2] = {};
    v8f accu[2][2] = {};

    float4 ra[4], rg[2], ru[2];
#pragma unroll
    for (int li = 0; li < 4; ++li) ra[li] = *(const float4*)(pA[li]);
#pragma unroll
    for (int li = 0; li < 2; ++li) { rg[li] = *(const float4*)(pBg[li]);
                                     ru[li] = *(const float4*)(pBu[li]); }

    for (int kt = 0; kt < H_DIM / 32; ++kt) {
        // ---- stage registers (tile kt) into LDS as bf16 ----
#pragma unroll
        for (int li = 0; li < 4; ++li) {
            unsigned* d = (unsigned*)&As[mA[li] * 36 + kkA[li]];
            d[0] = pack2bf(ra[li].x, ra[li].y);
            d[1] = pack2bf(ra[li].z, ra[li].w);
        }
#pragma unroll
        for (int li = 0; li < 2; ++li) {
            const int kk = kkB[li], n = nB[li];
            const unsigned g01 = pack2bf(rg[li].x, rg[li].y);
            const unsigned g23 = pack2bf(rg[li].z, rg[li].w);
            const unsigned u01 = pack2bf(ru[li].x, ru[li].y);
            const unsigned u23 = pack2bf(ru[li].z, ru[li].w);
            Bg[(n + 0) * 36 + kk] = (ushort_t)g01; Bg[(n + 1) * 36 + kk] = (ushort_t)(g01 >> 16);
            Bg[(n + 2) * 36 + kk] = (ushort_t)g23; Bg[(n + 3) * 36 + kk] = (ushort_t)(g23 >> 16);
            Bu[(n + 0) * 36 + kk] = (ushort_t)u01; Bu[(n + 1) * 36 + kk] = (ushort_t)(u01 >> 16);
            Bu[(n + 2) * 36 + kk] = (ushort_t)u23; Bu[(n + 3) * 36 + kk] = (ushort_t)(u23 >> 16);
        }
        __syncthreads();

        // ---- issue global loads for tile kt+1 (overlap with WMMA below) ----
        if (kt + 1 < H_DIM / 32) {
            const size_t ka = (size_t)(kt + 1) * 32;
#pragma unroll
            for (int li = 0; li < 4; ++li) ra[li] = *(const float4*)(pA[li] + ka);
#pragma unroll
            for (int li = 0; li < 2; ++li) { rg[li] = *(const float4*)(pBg[li] + ka * I_DIM);
                                             ru[li] = *(const float4*)(pBu[li] + ka * I_DIM); }
            if (kt + 2 < H_DIM / 32) {   // gfx1250 global_prefetch_b8, next-next tile
                __builtin_prefetch(pBg[0] + (size_t)(kt + 2) * 32 * I_DIM, 0, 3);
                __builtin_prefetch(pBu[0] + (size_t)(kt + 2) * 32 * I_DIM, 0, 3);
            }
        }

        // ---- WMMA on tile kt ----
        v16bf a[2], bg[2], bu[2];
#pragma unroll
        for (int tm = 0; tm < 2; ++tm) a[tm]  = load_a_frag(As, wm + tm * 16 + l16, half);
#pragma unroll
        for (int tn = 0; tn < 2; ++tn) { bg[tn] = load_b_frag(Bg, wn + tn * 16 + l16, half);
                                         bu[tn] = load_b_frag(Bu, wn + tn * 16 + l16, half); }
#pragma unroll
        for (int tm = 0; tm < 2; ++tm)
#pragma unroll
            for (int tn = 0; tn < 2; ++tn) {
                accg[tm][tn] = wmma_bf16(a[tm], bg[tn], accg[tm][tn]);
                accu[tm][tn] = wmma_bf16(a[tm], bu[tn], accu[tm][tn]);
            }
        __syncthreads();
    }

    // ---- epilogue: h = silu(g) * u, stored bf16 (down GEMM needs bf16 anyway)
#pragma unroll
    for (int tm = 0; tm < 2; ++tm)
#pragma unroll
        for (int tn = 0; tn < 2; ++tn)
#pragma unroll
            for (int r = 0; r < 8; ++r) {
                const int row = m0 + wm + tm * 16 + r + 8 * half;
                const int col = n0 + wn + tn * 16 + l16;
                if (row < rows) {
                    const float gv = accg[tm][tn][r];
                    const float uv = accu[tm][tn][r];
                    const float sv = gv / (1.0f + expf(-gv));
                    h_bf[(size_t)(rbase + row) * I_DIM + col] = f2bf(sv * uv);
                }
            }
}

// ============================================================================
// Kernel 4: down projection grouped GEMM: out = h @ down_w[g]  (K = 512)
// A is bf16; with async-to-LDS support the A tile is DMA'd (double buffered).
// ============================================================================
__global__ __launch_bounds__(256) void down_kernel(
    const ushort_t* __restrict__ h_bf,
    const float* __restrict__ down_w, const float* __restrict__ sdown_w,
    const int* __restrict__ row_base_g,
    float* __restrict__ out_buf) {
    const int g    = blockIdx.z;
    const int n0   = blockIdx.x * 64;
    const int m0   = blockIdx.y * 128;
    const int rows = row_base_g[E_NUM + 1 + g];
    if (m0 >= rows) return;
    const int rbase = row_base_g[g];
    const float* W = (g < E_NUM) ? down_w + (size_t)g * I_DIM * H_DIM : sdown_w;

    const int tid  = threadIdx.x;
    const int lane = tid & 31;
    const int wv   = tid >> 5;
    const int wm   = (wv & 3) * 32;
    const int wn   = (wv >> 2) * 32;
    const int half = lane >> 4;
    const int l16  = lane & 15;

    // ---- hoisted geometry; branchless A via row clamp ----
    int mA[2], kkA[2];
    const ushort_t* pA[2];
#pragma unroll
    for (int li = 0; li < 2; ++li) {
        const int idx = tid + li * 256;           // 0..511 => (m, 8-wide k chunk)
        mA[li]  = idx >> 2;
        kkA[li] = (idx & 3) * 8;
        int row = m0 + mA[li]; row = row < rows ? row : rows - 1;
        pA[li] = h_bf + (size_t)(rbase + row) * I_DIM + kkA[li];
    }
    int kkB[2], nB[2];
    const float* pB[2];
#pragma unroll
    for (int li = 0; li < 2; ++li) {
        const int idx = tid + li * 256;
        kkB[li] = idx >> 4;
        nB[li]  = (idx & 15) * 4;
        pB[li] = W + (size_t)kkB[li] * H_DIM + n0 + nB[li];
    }

    v8f acc[2][2] = {};
    float4 rb[2];
#pragma unroll
    for (int li = 0; li < 2; ++li) rb[li] = *(const float4*)(pB[li]);

#if HAVE_ASYNC_LDS
    // ================= async-to-LDS double-buffered A path =================
    __shared__ __align__(16) ushort_t As[2][128 * 36];
    __shared__ __align__(16) ushort_t Bs[64 * 36];

    auto issueA = [&](int kt, int buf) {
#pragma unroll
        for (int li = 0; li < 2; ++li) {
            const ushort_t* src = pA[li] + (size_t)kt * 32;
            ushort_t* dst = &As[buf][mA[li] * 36 + kkA[li]];
            __builtin_amdgcn_global_load_async_to_lds_b64(
                (v2i_t*)(uintptr_t)(const void*)src,
                (AS3 v2i_t*)(unsigned)(uintptr_t)(void*)dst, 0, 0);
            __builtin_amdgcn_global_load_async_to_lds_b64(
                (v2i_t*)(uintptr_t)(const void*)(src + 4),
                (AS3 v2i_t*)(unsigned)(uintptr_t)(void*)(dst + 4), 0, 0);
        }
    };
    issueA(0, 0);

    for (int kt = 0; kt < I_DIM / 32; ++kt) {
        const int cur = kt & 1;
#pragma unroll
        for (int li = 0; li < 2; ++li) {
            const int kk = kkB[li], n = nB[li];
            const unsigned b01 = pack2bf(rb[li].x, rb[li].y);
            const unsigned b23 = pack2bf(rb[li].z, rb[li].w);
            Bs[(n + 0) * 36 + kk] = (ushort_t)b01; Bs[(n + 1) * 36 + kk] = (ushort_t)(b01 >> 16);
            Bs[(n + 2) * 36 + kk] = (ushort_t)b23; Bs[(n + 3) * 36 + kk] = (ushort_t)(b23 >> 16);
        }
        __builtin_amdgcn_s_wait_asynccnt(0);   // own async copies landed in LDS
        __syncthreads();                       // -> all waves' copies visible

        if (kt + 1 < I_DIM / 32) {
            issueA(kt + 1, cur ^ 1);           // DMA next A tile into other buffer
            const size_t ka = (size_t)(kt + 1) * 32;
#pragma unroll
            for (int li = 0; li < 2; ++li) rb[li] = *(const float4*)(pB[li] + ka * H_DIM);
            if (kt + 2 < I_DIM / 32)
                __builtin_prefetch(pB[0] + (size_t)(kt + 2) * 32 * H_DIM, 0, 3);
        }

        v16bf a[2], b[2];
#pragma unroll
        for (int tm = 0; tm < 2; ++tm) a[tm] = load_a_frag(As[cur], wm + tm * 16 + l16, half);
#pragma unroll
        for (int tn = 0; tn < 2; ++tn) b[tn] = load_b_frag(Bs, wn + tn * 16 + l16, half);
#pragma unroll
        for (int tm = 0; tm < 2; ++tm)
#pragma unroll
            for (int tn = 0; tn < 2; ++tn)
                acc[tm][tn] = wmma_bf16(a[tm], b[tn], acc[tm][tn]);
        __syncthreads();
    }
#else
    // ================= register-staged fallback =================
    __shared__ __align__(16) ushort_t As[128 * 36];
    __shared__ __align__(16) ushort_t Bs[64 * 36];

    uint4 rha[2];
#pragma unroll
    for (int li = 0; li < 2; ++li) rha[li] = *(const uint4*)(pA[li]);

    for (int kt = 0; kt < I_DIM / 32; ++kt) {
#pragma unroll
        for (int li = 0; li < 2; ++li) {
            unsigned* d = (unsigned*)&As[mA[li] * 36 + kkA[li]];
            d[0] = rha[li].x; d[1] = rha[li].y; d[2] = rha[li].z; d[3] = rha[li].w;
            const int kk = kkB[li], n = nB[li];
            const unsigned b01 = pack2bf(rb[li].x, rb[li].y);
            const unsigned b23 = pack2bf(rb[li].z, rb[li].w);
            Bs[(n + 0) * 36 + kk] = (ushort_t)b01; Bs[(n + 1) * 36 + kk] = (ushort_t)(b01 >> 16);
            Bs[(n + 2) * 36 + kk] = (ushort_t)b23; Bs[(n + 3) * 36 + kk] = (ushort_t)(b23 >> 16);
        }
        __syncthreads();

        if (kt + 1 < I_DIM / 32) {
            const size_t ka = (size_t)(kt + 1) * 32;
#pragma unroll
            for (int li = 0; li < 2; ++li) { rha[li] = *(const uint4*)(pA[li] + ka);
                                             rb[li]  = *(const float4*)(pB[li] + ka * H_DIM); }
            if (kt + 2 < I_DIM / 32)
                __builtin_prefetch(pB[0] + (size_t)(kt + 2) * 32 * H_DIM, 0, 3);
        }

        v16bf a[2], b[2];
#pragma unroll
        for (int tm = 0; tm < 2; ++tm) a[tm] = load_a_frag(As, wm + tm * 16 + l16, half);
#pragma unroll
        for (int tn = 0; tn < 2; ++tn) b[tn] = load_b_frag(Bs, wn + tn * 16 + l16, half);
#pragma unroll
        for (int tm = 0; tm < 2; ++tm)
#pragma unroll
            for (int tn = 0; tn < 2; ++tn)
                acc[tm][tn] = wmma_bf16(a[tm], b[tn], acc[tm][tn]);
        __syncthreads();
    }
#endif

#pragma unroll
    for (int tm = 0; tm < 2; ++tm)
#pragma unroll
        for (int tn = 0; tn < 2; ++tn)
#pragma unroll
            for (int r = 0; r < 8; ++r) {
                const int row = m0 + wm + tm * 16 + r + 8 * half;
                const int col = n0 + wn + tn * 16 + l16;
                if (row < rows)
                    out_buf[(size_t)(rbase + row) * H_DIM + col] = acc[tm][tn][r];
            }
}

// ============================================================================
// Kernel 5: combine — fixed k-order weighted sum + shared expert (deterministic)
// ============================================================================
__global__ void combine_kernel(const float* __restrict__ out_buf,
                               const int* __restrict__ topk_idx,
                               const float* __restrict__ topk_w,
                               const int* __restrict__ assign_slot,
                               const int* __restrict__ row_base_g,
                               float* __restrict__ out) {
    const int t  = blockIdx.x;
    const int hh = threadIdx.x * 4;        // 256 threads * 4 = H
    float4 acc = make_float4(0.f, 0.f, 0.f, 0.f);
#pragma unroll
    for (int k = 0; k < K_TOP; ++k) {
        const int a = t * K_TOP + k;
        const int s = assign_slot[a];
        if (s >= 0) {
            const int e = topk_idx[a];
            const float w = topk_w[a];
            const float4 v = *(const float4*)(out_buf + (size_t)(row_base_g[e] + s) * H_DIM + hh);
            acc.x += w * v.x; acc.y += w * v.y; acc.z += w * v.z; acc.w += w * v.w;
        }
    }
    const float4 sv = *(const float4*)(out_buf + (size_t)(row_base_g[E_NUM] + t) * H_DIM + hh);
    acc.x += sv.x; acc.y += sv.y; acc.z += sv.z; acc.w += sv.w;
    *(float4*)(out + (size_t)t * H_DIM + hh) = acc;
}

// ============================================================================
extern "C" void kernel_launch(void* const* d_in, const int* in_sizes, int n_in,
                              void* d_out, int out_size, void* d_ws, size_t ws_size,
                              hipStream_t stream) {
    const float* x    = (const float*)d_in[0];
    const float* rw   = (const float*)d_in[1];
    const float* gw   = (const float*)d_in[2];
    const float* uw   = (const float*)d_in[3];
    const float* dw   = (const float*)d_in[4];
    const float* sgw  = (const float*)d_in[5];
    const float* suw  = (const float*)d_in[6];
    const float* sdw  = (const float*)d_in[7];
    float* out = (float*)d_out;

    char* ws = (char*)d_ws;
    size_t off = 0;
    auto alloc = [&](size_t bytes) -> void* {
        off = (off + 255) & ~(size_t)255;
        void* p = ws + off;
        off += bytes;
        return p;
    };
    int*      topk_idx    = (int*)     alloc((size_t)T_TOK * K_TOP * 4);
    float*    topk_w      = (float*)   alloc((size_t)T_TOK * K_TOP * 4);
    int*      assign_slot = (int*)     alloc((size_t)T_TOK * K_TOP * 4);
    int*      row_base_g  = (int*)     alloc((size_t)2 * (E_NUM + 2) * 4);
    int*      tok_list    = (int*)     alloc((size_t)ROWS_MAX * 4);
    ushort_t* h_bf        = (ushort_t*)alloc((size_t)ROWS_MAX * I_DIM * 2);
    float*    out_buf     = (float*)   alloc((size_t)ROWS_MAX * H_DIM * 4);
    (void)ws_size; (void)in_sizes; (void)n_in; (void)out_size;

    router_kernel<<<T_TOK, 64, 0, stream>>>(x, rw, topk_idx, topk_w);
    dispatch_kernel<<<1, 128, 0, stream>>>(topk_idx, assign_slot, tok_list, row_base_g);
    gateup_kernel<<<dim3(I_DIM / 64, (T_TOK + 127) / 128, E_NUM + 1), 256, 0, stream>>>(
        x, gw, uw, sgw, suw, tok_list, row_base_g, h_bf);
    down_kernel<<<dim3(H_DIM / 64, (T_TOK + 127) / 128, E_NUM + 1), 256, 0, stream>>>(
        h_bf, dw, sdw, row_base_g, out_buf);
    combine_kernel<<<T_TOK, 256, 0, stream>>>(out_buf, topk_idx, topk_w,
                                              assign_slot, row_base_g, out);
}
